// DecoderBlock_64630667870546
// MI455X (gfx1250) — matrix operand order repaired and checked by reading, not compiled, not run on hardware
//
#include <hip/hip_runtime.h>

// ---------------------------------------------------------------------------
// CDNA5 (gfx1250) decoder block: LN -> QKV -> causal flash attention -> Wo
// -> +res -> LN -> GELU MLP -> +res.
// Matmuls: v_wmma_f32_16x16x32_bf16.  Tile staging: global_load_async_to_lds
// with double-buffered LDS tiles (prefetch tile t+1 while computing tile t).
// ---------------------------------------------------------------------------

typedef __attribute__((ext_vector_type(8)))  __bf16 v8bf;
typedef __attribute__((ext_vector_type(16))) __bf16 v16bf;
typedef __attribute__((ext_vector_type(8)))  float  v8f;

__device__ __forceinline__ v16bf ld_frag2(const __bf16* p0, const __bf16* p1) {
  union { v16bf v; v8bf h[2]; } u;
  u.h[0] = *(const v8bf*)p0;
  u.h[1] = *(const v8bf*)p1;
  return u.v;
}

__device__ __forceinline__ v8f wmma_bf16(v16bf a, v16bf b, v8f c) {
  // D = A(16x32 bf16) * B(32x16 bf16) + C(16x16 f32)
  return __builtin_amdgcn_wmma_f32_16x16x32_bf16(false, a, false, b,
                                                 (short)0, c, false, false);
}

// Per-lane async 16B copy: global memory -> LDS (tracked by ASYNCcnt).
// GV addressing: 64-bit vaddr pair, saddr=off.  vdst holds the LDS byte
// address (low 32 bits of the generic __shared__ pointer == LDS offset).
__device__ __forceinline__ void async_b128(void* lds, const void* gptr) {
  unsigned lo = (unsigned)(unsigned long long)lds;
  unsigned long long ga = (unsigned long long)gptr;
  asm volatile("global_load_async_to_lds_b128 %0, %1, off"
               :: "v"(lo), "v"(ga) : "memory");
}
__device__ __forceinline__ void wait_async0() {
  asm volatile("s_wait_asynccnt 0x0" ::: "memory");
}

// ---------------------------------------------------------------------------
// fp32 -> bf16 conversion
// ---------------------------------------------------------------------------
__global__ void tb_cvt_f32_bf16(const float* __restrict__ in,
                                __bf16* __restrict__ out, long n) {
  long i = ((long)blockIdx.x * blockDim.x + threadIdx.x) * 4;
  if (i + 3 < n) {
    float4 f = *(const float4*)&in[i];
    out[i + 0] = (__bf16)f.x;
    out[i + 1] = (__bf16)f.y;
    out[i + 2] = (__bf16)f.z;
    out[i + 3] = (__bf16)f.w;
  } else {
    for (; i < n; ++i) out[i] = (__bf16)in[i];
  }
}

// ---------------------------------------------------------------------------
// LayerNorm over D=1024, one block per row, bf16 output
// ---------------------------------------------------------------------------
__global__ __launch_bounds__(256)
void tb_layernorm(const float* __restrict__ x, const float* __restrict__ gamma,
                  const float* __restrict__ beta, __bf16* __restrict__ out) {
  const int D = 1024;
  int row = blockIdx.x;
  const float* xr = x + (long)row * D;
  int tid = threadIdx.x;
  float v[4], s = 0.f, s2 = 0.f;
#pragma unroll
  for (int i = 0; i < 4; ++i) {
    v[i] = xr[tid + 256 * i];
    s += v[i];
    s2 += v[i] * v[i];
  }
#pragma unroll
  for (int off = 16; off; off >>= 1) {
    s  += __shfl_xor(s,  off, 32);
    s2 += __shfl_xor(s2, off, 32);
  }
  __shared__ float ss[8], ss2[8];
  int wave = tid >> 5, lane = tid & 31;
  if (lane == 0) { ss[wave] = s; ss2[wave] = s2; }
  __syncthreads();
  if (wave == 0) {
    s  = (lane < 8) ? ss[lane]  : 0.f;
    s2 = (lane < 8) ? ss2[lane] : 0.f;
#pragma unroll
    for (int off = 4; off; off >>= 1) {
      s  += __shfl_xor(s,  off, 32);
      s2 += __shfl_xor(s2, off, 32);
    }
    if (lane == 0) { ss[0] = s; ss2[0] = s2; }
  }
  __syncthreads();
  float mu  = ss[0] * (1.f / D);
  float var = ss2[0] * (1.f / D) - mu * mu;
  float inv = rsqrtf(var + 1e-5f);
  __bf16* orow = out + (long)row * D;
#pragma unroll
  for (int i = 0; i < 4; ++i) {
    int c = tid + 256 * i;
    orow[c] = (__bf16)((v[i] - mu) * inv * gamma[c] + beta[c]);
  }
}

// ---------------------------------------------------------------------------
// Strided-batched bf16 WMMA GEMM:  C[z] = epilogue(A[z] @ B[z])
// Block tile 64x64, 4 waves, wave = 16 rows x 64 cols (4 wmma tiles).
// Double-buffered LDS: A tiles via async-to-LDS, B tiles held in registers
// during compute and scattered transposed ([n][k]) afterwards.
// M%64==0, N%64==0, K%32==0 (true for all uses here).
// ---------------------------------------------------------------------------
template <bool HAS_BIAS, bool DO_GELU, bool HAS_RESID, bool OUT_F32, bool OUT_BF16>
__global__ __launch_bounds__(128)
void tb_gemm_bf16(const __bf16* __restrict__ A, const __bf16* __restrict__ Bw,
                  const float* __restrict__ bias, const float* __restrict__ resid,
                  float* __restrict__ Cf, __bf16* __restrict__ Cb,
                  int M, int N, int K, int lda, int ldb, int ldc,
                  long sA, long sB, long sBias, long sC) {
  long z = blockIdx.z;
  A  += z * sA;
  Bw += z * sB;
  const float* biasz = HAS_BIAS ? bias + z * sBias : nullptr;
  long coff = z * sC;

  __shared__ __bf16 As[2][64][32];   // [buf][m][k]
  __shared__ __bf16 Bs[2][64][32];   // [buf][n][k]

  int tid = threadIdx.x;
  int wave = tid >> 5, lane = tid & 31;
  int half = lane >> 4, l16 = lane & 15;
  int m0 = wave * 16;
  long mBase = (long)blockIdx.y * 64;
  long nBase = (long)blockIdx.x * 64;

  // per-thread stationary tile-fill coordinates
  int ar0 = (tid * 8) >> 5,         ac0 = (tid * 8) & 31;          // A it=0
  int ar1 = ((128 + tid) * 8) >> 5, ac1 = ((128 + tid) * 8) & 31;  // A it=1
  int bk0 = (tid * 8) >> 6,         bn0 = (tid * 8) & 63;          // B it=0
  int bk1 = ((128 + tid) * 8) >> 6, bn1 = ((128 + tid) * 8) & 63;  // B it=1

  v8bf bg0, bg1;
  // ---- prologue: stage tile 0 ----
  async_b128(&As[0][ar0][ac0], &A[(mBase + ar0) * (long)lda + ac0]);
  async_b128(&As[0][ar1][ac1], &A[(mBase + ar1) * (long)lda + ac1]);
  bg0 = *(const v8bf*)&Bw[(long)bk0 * ldb + nBase + bn0];
  bg1 = *(const v8bf*)&Bw[(long)bk1 * ldb + nBase + bn1];
#pragma unroll
  for (int j = 0; j < 8; ++j) {
    Bs[0][bn0 + j][bk0] = bg0[j];
    Bs[0][bn1 + j][bk1] = bg1[j];
  }
  wait_async0();
  __syncthreads();

  v8f acc[4] = {};
  int nk = K >> 5;
  for (int kt = 0; kt < nk; ++kt) {
    int cur = kt & 1, nxt = cur ^ 1;
    bool pf = (kt + 1) < nk;
    if (pf) {
      int k0 = (kt + 1) << 5;
      async_b128(&As[nxt][ar0][ac0], &A[(mBase + ar0) * (long)lda + k0 + ac0]);
      async_b128(&As[nxt][ar1][ac1], &A[(mBase + ar1) * (long)lda + k0 + ac1]);
      bg0 = *(const v8bf*)&Bw[(long)(k0 + bk0) * ldb + nBase + bn0];
      bg1 = *(const v8bf*)&Bw[(long)(k0 + bk1) * ldb + nBase + bn1];
    }

    // fragments first (give the scheduler slack), then the wmma chain
    v16bf af = ld_frag2(&As[cur][m0 + l16][half * 8],
                        &As[cur][m0 + l16][16 + half * 8]);
    v16bf bfr[4];
#pragma unroll
    for (int t = 0; t < 4; ++t)
      bfr[t] = ld_frag2(&Bs[cur][t * 16 + l16][half * 8],
                        &Bs[cur][t * 16 + l16][16 + half * 8]);
#pragma unroll
    for (int t = 0; t < 4; ++t) acc[t] = wmma_bf16(af, bfr[t], acc[t]);

    if (pf) {
#pragma unroll
      for (int j = 0; j < 8; ++j) {
        Bs[nxt][bn0 + j][bk0] = bg0[j];
        Bs[nxt][bn1 + j][bk1] = bg1[j];
      }
    }
    wait_async0();
    __syncthreads();
  }

  // Epilogue.  C layout: row = r + 8*half, col = l16 (per 16x16 tile)
#pragma unroll
  for (int t = 0; t < 4; ++t) {
    long ncol = nBase + t * 16 + l16;
    float bv = HAS_BIAS ? biasz[ncol] : 0.f;
#pragma unroll
    for (int r = 0; r < 8; ++r) {
      long mrow = mBase + m0 + r + 8 * half;
      float v = acc[t][r] + bv;
      if (DO_GELU) v = 0.5f * v * (1.0f + erff(v * 0.70710678118f));
      if (HAS_RESID) v += resid[mrow * (long)ldc + ncol];
      long ci = coff + mrow * (long)ldc + ncol;
      if (OUT_F32)  Cf[ci] = v;
      if (OUT_BF16) Cb[ci] = (__bf16)v;
    }
  }
}

// ---------------------------------------------------------------------------
// Causal flash attention per (b, head).  Q/K/V bf16 laid out [H, B*S, 64].
// Block = 64 query rows, 4 waves x 16 rows; online softmax.
// Q/K tiles via async-to-LDS; K/V double-buffered across the key loop
// (prefetch next key block during softmax + P@V of the current one).
// Output written merged to [B*S, 1024] at column h*64.
// ---------------------------------------------------------------------------
__global__ __launch_bounds__(128)
void tb_attention(const __bf16* __restrict__ Q, const __bf16* __restrict__ Km,
                  const __bf16* __restrict__ Vm, __bf16* __restrict__ O) {
  const int S = 2048, HD = 64, Bn = 4, H = 16, Dm = 1024;
  int qb = blockIdx.x;                    // query block (64 rows)
  int bh = blockIdx.y;
  int b = bh / H, hh = bh % H;
  long base = ((long)hh * Bn * S + (long)b * S) * HD;
  const __bf16* Qp = Q + base;
  const __bf16* Kp = Km + base;
  const __bf16* Vp = Vm + base;

  __shared__ __bf16 Qt[64][64];          // [q][hd]
  __shared__ __bf16 Kt[2][64][64];       // [buf][key][hd]
  __shared__ __bf16 VtT[2][64][64];      // [buf][hd][key]
  __shared__ __bf16 Pt[4][16][64];       // per-wave probability tile

  int tid = threadIdx.x, wave = tid >> 5, lane = tid & 31;
  int half = lane >> 4, l16 = lane & 15;
  int m0 = wave * 16;
  int qbase = qb * 64;

  // stationary fill coordinates: 4 chunks of 8 bf16 per thread
  int fr[4], fc[4];
#pragma unroll
  for (int it = 0; it < 4; ++it) {
    int idx = (it * 128 + tid) * 8;
    fr[it] = idx >> 6;
    fc[it] = idx & 63;
  }

  // ---- prologue: stage Q tile and key-block 0 (buffer 0) ----
  v8bf vg[4];
#pragma unroll
  for (int it = 0; it < 4; ++it) {
    async_b128(&Qt[fr[it]][fc[it]], &Qp[(long)(qbase + fr[it]) * HD + fc[it]]);
    async_b128(&Kt[0][fr[it]][fc[it]], &Kp[(long)fr[it] * HD + fc[it]]);
    vg[it] = *(const v8bf*)&Vp[(long)fr[it] * HD + fc[it]];
  }
#pragma unroll
  for (int it = 0; it < 4; ++it)
#pragma unroll
    for (int j = 0; j < 8; ++j) VtT[0][fc[it] + j][fr[it]] = vg[it][j];
  wait_async0();
  __syncthreads();

  v16bf qa[2];
#pragma unroll
  for (int c = 0; c < 2; ++c)
    qa[c] = ld_frag2(&Qt[m0 + l16][32 * c + half * 8],
                     &Qt[m0 + l16][32 * c + 16 + half * 8]);

  v8f oacc[4] = {};
  float mrun[8], lrun[8];
#pragma unroll
  for (int r = 0; r < 8; ++r) { mrun[r] = -3.0e38f; lrun[r] = 0.f; }
  const float LOG2E = 1.44269504f;

  for (int kb = 0; kb <= qb; ++kb) {
    int kbase = kb * 64;
    int cur = kb & 1, nxt = cur ^ 1;
    bool pf = (kb + 1) <= qb;
    if (pf) {
      int nb = (kb + 1) * 64;
#pragma unroll
      for (int it = 0; it < 4; ++it) {
        async_b128(&Kt[nxt][fr[it]][fc[it]],
                   &Kp[(long)(nb + fr[it]) * HD + fc[it]]);
        vg[it] = *(const v8bf*)&Vp[(long)(nb + fr[it]) * HD + fc[it]];
      }
    }

    // scores = Q @ K^T: load all K fragments, then the wmma chain
    v16bf kf[4][2];
#pragma unroll
    for (int t = 0; t < 4; ++t) {
      kf[t][0] = ld_frag2(&Kt[cur][t * 16 + l16][half * 8],
                          &Kt[cur][t * 16 + l16][16 + half * 8]);
      kf[t][1] = ld_frag2(&Kt[cur][t * 16 + l16][32 + half * 8],
                          &Kt[cur][t * 16 + l16][48 + half * 8]);
    }
    v8f st[4];
#pragma unroll
    for (int t = 0; t < 4; ++t) {
      v8f zacc = {};
      zacc = wmma_bf16(qa[0], kf[t][0], zacc);
      zacc = wmma_bf16(qa[1], kf[t][1], zacc);
      st[t] = zacc;
    }

    // scale, causal mask, online softmax (row = r + 8*half, 16 lanes/row)
#pragma unroll
    for (int r = 0; r < 8; ++r) {
      int rowg = qbase + m0 + r + 8 * half;
      float sv[4];
      float mloc = -3.0e38f;
#pragma unroll
      for (int t = 0; t < 4; ++t) {
        int colg = kbase + t * 16 + l16;
        float xv = st[t][r] * 0.125f;             // 1/sqrt(64)
        if (colg > rowg) xv = -3.0e38f;
        sv[t] = xv;
        mloc = fmaxf(mloc, xv);
      }
#pragma unroll
      for (int off = 1; off < 16; off <<= 1)
        mloc = fmaxf(mloc, __shfl_xor(mloc, off, 32));
      float mnew = fmaxf(mrun[r], mloc);
      float corr = exp2f((mrun[r] - mnew) * LOG2E);
      float lloc = 0.f;
#pragma unroll
      for (int t = 0; t < 4; ++t) {
        float pv = exp2f((sv[t] - mnew) * LOG2E);
        lloc += pv;
        Pt[wave][r + 8 * half][t * 16 + l16] = (__bf16)pv;
      }
#pragma unroll
      for (int off = 1; off < 16; off <<= 1) lloc += __shfl_xor(lloc, off, 32);
      lrun[r] = lrun[r] * corr + lloc;
      mrun[r] = mnew;
#pragma unroll
      for (int t = 0; t < 4; ++t) oacc[t][r] *= corr;
    }
    asm volatile("s_wait_dscnt 0x0" ::: "memory");  // Pt write->read, same wave

    // O += P @ V: load all fragments, then the wmma chain
    v16bf pa0 = ld_frag2(&Pt[wave][l16][half * 8],
                         &Pt[wave][l16][16 + half * 8]);
    v16bf pa1 = ld_frag2(&Pt[wave][l16][32 + half * 8],
                         &Pt[wave][l16][48 + half * 8]);
    v16bf vf[4][2];
#pragma unroll
    for (int t = 0; t < 4; ++t) {
      vf[t][0] = ld_frag2(&VtT[cur][t * 16 + l16][half * 8],
                          &VtT[cur][t * 16 + l16][16 + half * 8]);
      vf[t][1] = ld_frag2(&VtT[cur][t * 16 + l16][32 + half * 8],
                          &VtT[cur][t * 16 + l16][48 + half * 8]);
    }
#pragma unroll
    for (int t = 0; t < 4; ++t) {
      oacc[t] = wmma_bf16(pa0, vf[t][0], oacc[t]);
      oacc[t] = wmma_bf16(pa1, vf[t][1], oacc[t]);
    }

    if (pf) {
#pragma unroll
      for (int it = 0; it < 4; ++it)
#pragma unroll
        for (int j = 0; j < 8; ++j) VtT[nxt][fc[it] + j][fr[it]] = vg[it][j];
    }
    wait_async0();
    __syncthreads();
  }

  // normalize and scatter to merged-head layout [B*S, 1024]
#pragma unroll
  for (int r = 0; r < 8; ++r) {
    float inv = (lrun[r] > 0.f) ? 1.f / lrun[r] : 0.f;
    long rowg = qbase + m0 + r + 8 * half;
    long obase = ((long)b * S + rowg) * Dm + hh * 64;
#pragma unroll
    for (int t = 0; t < 4; ++t)
      O[obase + t * 16 + l16] = (__bf16)(oacc[t][r] * inv);
  }
}

// ---------------------------------------------------------------------------
// Host orchestration
// ---------------------------------------------------------------------------
static inline void cvt(const float* in, __bf16* out, long n, hipStream_t s) {
  long blocks = (n + 1023) / 1024;
  tb_cvt_f32_bf16<<<dim3((unsigned)blocks), 256, 0, s>>>(in, out, n);
}

extern "C" void kernel_launch(void* const* d_in, const int* in_sizes, int n_in,
                              void* d_out, int out_size, void* d_ws, size_t ws_size,
                              hipStream_t stream) {
  (void)in_sizes; (void)n_in; (void)out_size; (void)ws_size;
  const int  Bn = 4, S = 2048, D = 1024, H = 16, HD = 64;
  const long BS = (long)Bn * S;             // 8192 rows

  const float* x     = (const float*)d_in[0];
  const float* Wq    = (const float*)d_in[1];
  const float* Wk    = (const float*)d_in[2];
  const float* Wv    = (const float*)d_in[3];
  const float* bq    = (const float*)d_in[4];
  const float* bk    = (const float*)d_in[5];
  const float* bv    = (const float*)d_in[6];
  const float* Wo    = (const float*)d_in[7];
  const float* bo    = (const float*)d_in[8];
  const float* W1    = (const float*)d_in[9];
  const float* b1    = (const float*)d_in[10];
  const float* W2    = (const float*)d_in[11];
  const float* b2    = (const float*)d_in[12];
  const float* gamma = (const float*)d_in[13];
  const float* beta  = (const float*)d_in[14];
  float* out = (float*)d_out;

  // workspace carving (256B aligned), total ~216 MB
  char* p = (char*)d_ws;
  auto carve = [&](size_t bytes) -> char* {
    char* r = p;
    p += (bytes + 255) & ~(size_t)255;
    return r;
  };
  __bf16* hbf    = (__bf16*)carve(BS * D * 2);              // LN1 out
  __bf16* wq_bf  = (__bf16*)carve((size_t)H * D * HD * 2);
  __bf16* wk_bf  = (__bf16*)carve((size_t)H * D * HD * 2);
  __bf16* wv_bf  = (__bf16*)carve((size_t)H * D * HD * 2);
  __bf16* wo_bf  = (__bf16*)carve((size_t)D * D * 2);
  __bf16* w1_bf  = (__bf16*)carve((size_t)D * 4 * D * 2);
  __bf16* w2_bf  = (__bf16*)carve((size_t)4 * D * D * 2);
  __bf16* qbf    = (__bf16*)carve((size_t)H * BS * HD * 2); // [H, B*S, 64]
  __bf16* kbf    = (__bf16*)carve((size_t)H * BS * HD * 2);
  __bf16* vbf    = (__bf16*)carve((size_t)H * BS * HD * 2);
  __bf16* attnbf = (__bf16*)carve(BS * D * 2);              // merged heads
  float*  x1     = (float*)carve(BS * D * 4);               // post-attn resid
  __bf16* h2bf   = (__bf16*)carve(BS * D * 2);              // LN2 out
  __bf16* mbf    = (__bf16*)carve(BS * 4 * D * 2);          // MLP hidden

  // 1. weight conversions
  cvt(Wq, wq_bf, (long)H * D * HD, stream);
  cvt(Wk, wk_bf, (long)H * D * HD, stream);
  cvt(Wv, wv_bf, (long)H * D * HD, stream);
  cvt(Wo, wo_bf, (long)D * D, stream);
  cvt(W1, w1_bf, (long)D * 4 * D, stream);
  cvt(W2, w2_bf, (long)4 * D * D, stream);

  // 2. LN1
  tb_layernorm<<<dim3((unsigned)BS), 256, 0, stream>>>(x, gamma, beta, hbf);

  // 3. QKV projections: per-head batched GEMM [BS,1024]x[1024,64] -> bf16
  tb_gemm_bf16<true, false, false, false, true>
      <<<dim3(1, BS / 64, H), 128, 0, stream>>>(
      hbf, wq_bf, bq, nullptr, nullptr, qbf,
      (int)BS, HD, D, D, HD, HD, 0, (long)D * HD, HD, BS * HD);
  tb_gemm_bf16<true, false, false, false, true>
      <<<dim3(1, BS / 64, H), 128, 0, stream>>>(
      hbf, wk_bf, bk, nullptr, nullptr, kbf,
      (int)BS, HD, D, D, HD, HD, 0, (long)D * HD, HD, BS * HD);
  tb_gemm_bf16<true, false, false, false, true>
      <<<dim3(1, BS / 64, H), 128, 0, stream>>>(
      hbf, wv_bf, bv, nullptr, nullptr, vbf,
      (int)BS, HD, D, D, HD, HD, 0, (long)D * HD, HD, BS * HD);

  // 4. causal flash attention, output pre-merged to [BS, 1024]
  tb_attention<<<dim3(S / 64, Bn * H), 128, 0, stream>>>(qbf, kbf, vbf, attnbf);

  // 5. output projection + residual: x1 = x + attn @ Wo + bo  (f32 out)
  tb_gemm_bf16<true, false, true, true, false>
      <<<dim3(D / 64, BS / 64, 1), 128, 0, stream>>>(
      attnbf, wo_bf, bo, x, x1, nullptr,
      (int)BS, D, D, D, D, D, 0, 0, 0, 0);

  // 6. LN2
  tb_layernorm<<<dim3((unsigned)BS), 256, 0, stream>>>(x1, gamma, beta, h2bf);

  // 7. MLP up + exact GELU -> bf16
  tb_gemm_bf16<true, true, false, false, true>
      <<<dim3(4 * D / 64, BS / 64, 1), 128, 0, stream>>>(
      h2bf, w1_bf, b1, nullptr, nullptr, mbf,
      (int)BS, 4 * D, D, D, 4 * D, 4 * D, 0, 0, 0, 0);

  // 8. MLP down + bias + residual -> final f32 output
  tb_gemm_bf16<true, false, true, true, false>
      <<<dim3(D / 64, BS / 64, 1), 128, 0, stream>>>(
      mbf, w2_bf, b2, x1, out, nullptr,
      (int)BS, D, 4 * D, 4 * D, D, D, 0, 0, 0, 0);
}
